// SelfBlock_36060545417849
// MI455X (gfx1250) — matrix operand order repaired and checked by reading, not compiled, not run on hardware
//
#include <hip/hip_runtime.h>

// ---------------------------------------------------------------- constants
constexpr int Bb = 2;
constexpr int Nn = 2048;
constexpr int Cc = 1024;
constexpr int Hh = 16;
constexpr int Dd = 64;
constexpr int Tt = Bb * Nn;          // 4096 tokens

// ---------------------------------------------------------------- WMMA types
typedef __attribute__((ext_vector_type(16))) __bf16 v16bf;
typedef __attribute__((ext_vector_type(8)))  __bf16 v8bf;
typedef __attribute__((ext_vector_type(8)))  float  v8f;

#define WMMA_BF16(a, b, c) \
  __builtin_amdgcn_wmma_f32_16x16x32_bf16(false, (a), false, (b), (short)0, (c), false, false)

__device__ __forceinline__ v16bf cat8(v8bf lo, v8bf hi) {
  return __builtin_shufflevector(lo, hi, 0, 1, 2, 3, 4, 5, 6, 7,
                                 8, 9, 10, 11, 12, 13, 14, 15);
}

// A-fragment (16x32 bf16): lane L<16 -> row L, K = 0..15;
// lane L>=16 -> row L-16, K = 16..31. p is the per-lane base.
__device__ __forceinline__ v16bf load_afrag(const __bf16* p) {
  v8bf lo = *(const v8bf*)p;          // K = kb .. kb+7
  v8bf hi = *(const v8bf*)(p + 8);    // K = kb+8 .. kb+15
  return cat8(lo, hi);
}

// B-fragment (32x16 bf16): lane L<16 -> col L, K=0..15; lane L>=16 -> col L-16, K=16..31.
__device__ __forceinline__ v16bf load_bfrag(const __bf16* p) {
  v8bf lo = *(const v8bf*)p;
  v8bf hi = *(const v8bf*)(p + 8);
  return cat8(lo, hi);
}

// LDS tile helpers, row stride 48 bf16 (96B, 16B aligned, conflict-skewed)
__device__ __forceinline__ v16bf frag_a_lds(const __bf16* tile, int lane) {
  return load_afrag(tile + (lane & 15) * 48 + ((lane >> 4) << 4));
}
__device__ __forceinline__ v16bf frag_b_lds(const __bf16* tile, int lane) {
  return load_bfrag(tile + (lane & 15) * 48 + ((lane >> 4) << 4));
}

// ------------------------------------------------- async global->LDS staging
// Copies one 64B row-slab (32 bf16) from global to LDS. On gfx1250 this uses
// the ASYNCcnt-tracked async-DMA path (no VGPR round trip); otherwise falls
// back to b128 register staging.
#if __has_builtin(__builtin_amdgcn_global_load_async_to_lds_b128)
#define ASYNC_STAGE 1
#else
#define ASYNC_STAGE 0
#endif

#if ASYNC_STAGE
typedef int vint4 __attribute__((__vector_size__(16)));
typedef __attribute__((address_space(1))) vint4* gptr4;   // global int4*
typedef __attribute__((address_space(3))) vint4* lptr4;   // LDS int4*
#endif

__device__ __forceinline__ void stage_slab(const __bf16* g, __bf16* l) {
#if ASYNC_STAGE
  __builtin_amdgcn_global_load_async_to_lds_b128((gptr4)(g),      (lptr4)(l),      0, 0);
  __builtin_amdgcn_global_load_async_to_lds_b128((gptr4)(g + 8),  (lptr4)(l + 8),  0, 0);
  __builtin_amdgcn_global_load_async_to_lds_b128((gptr4)(g + 16), (lptr4)(l + 16), 0, 0);
  __builtin_amdgcn_global_load_async_to_lds_b128((gptr4)(g + 24), (lptr4)(l + 24), 0, 0);
#else
  const uint4* s4 = (const uint4*)g;
  uint4* d4 = (uint4*)l;
  uint4 t0 = s4[0], t1 = s4[1], t2 = s4[2], t3 = s4[3];
  d4[0] = t0; d4[1] = t1; d4[2] = t2; d4[3] = t3;
#endif
}

__device__ __forceinline__ void wait_stage() {
#if ASYNC_STAGE
#if __has_builtin(__builtin_amdgcn_s_wait_asynccnt)
  __builtin_amdgcn_s_wait_asynccnt(0);
#else
  asm volatile("s_wait_asynccnt 0x0" ::: "memory");
#endif
#endif
}

// ---------------------------------------------------------------- casts
__global__ void k_cast(const float* __restrict__ s, __bf16* __restrict__ d, int n) {
  for (int i = blockIdx.x * blockDim.x + threadIdx.x; i < n; i += gridDim.x * blockDim.x)
    d[i] = (__bf16)s[i];
}

// x -> x_bf and also into the first C columns of cc (T x 2C)
__global__ void k_cast_x(const float* __restrict__ x, __bf16* __restrict__ xb,
                         __bf16* __restrict__ cc, int n) {
  for (int i = blockIdx.x * blockDim.x + threadIdx.x; i < n; i += gridDim.x * blockDim.x) {
    float v = x[i];
    __bf16 bv = (__bf16)v;
    xb[i] = bv;
    int row = i >> 10;          // / C
    int col = i & (Cc - 1);     // % C
    cc[(size_t)row * (2 * Cc) + col] = bv;
  }
}

// ---------------------------------------------------------------- generic bf16 GEMM
// D[M,N] = A[M,K] * W[N,K]^T + bias ; optional f32 out, bf16 out, residual add.
// Block: 128 threads (4 waves), 64x64 tile; wave -> 32x32 (2x2 WMMA tiles).
// Double-buffered LDS: async-DMA the next K-slice while WMMAs run on the
// current one; one s_wait_asynccnt + barrier per iteration.
__global__ __launch_bounds__(128) void k_gemm(
    const __bf16* __restrict__ A, int lda,
    const __bf16* __restrict__ W, int ldw,
    const float* __restrict__ bias, int K,
    float* __restrict__ outF, __bf16* __restrict__ outB, int ldo,
    const float* __restrict__ resid, int ldr) {
  __shared__ __bf16 lA[2][64 * 48];
  __shared__ __bf16 lW[2][64 * 48];
  const int tid  = threadIdx.x;
  const int lane = tid & 31;
  const int wave = tid >> 5;
  const int wm = (wave >> 1) * 32;
  const int wn = (wave & 1) * 32;
  const size_t mBlock = (size_t)blockIdx.y * 64;
  const size_t nBlock = (size_t)blockIdx.x * 64;

  v8f acc[2][2] = {};

  const int r0 = tid & 63;
  const __bf16* gsrc = (tid < 64) ? (A + (mBlock + r0) * (size_t)lda)
                                  : (W + (nBlock + r0) * (size_t)ldw);
  __bf16* ldst[2] = { ((tid < 64) ? lA[0] : lW[0]) + r0 * 48,
                      ((tid < 64) ? lA[1] : lW[1]) + r0 * 48 };

  // prime buffer 0 with the first K-slice
  stage_slab(gsrc, ldst[0]);
  wait_stage();
  __syncthreads();

  int buf = 0;
  for (int k0 = 0; k0 < K; k0 += 32) {
    // kick off DMA of the next slice into the other buffer
    if (k0 + 32 < K) {
      stage_slab(gsrc + k0 + 32, ldst[buf ^ 1]);
      if (k0 + 64 < K) __builtin_prefetch(gsrc + k0 + 64, 0, 1);  // warm L2
    }

    const __bf16* tA = lA[buf];
    const __bf16* tW = lW[buf];
#pragma unroll
    for (int im = 0; im < 2; ++im) {
      v16bf af = frag_a_lds(tA + (wm + im * 16) * 48, lane);
#pragma unroll
      for (int jn = 0; jn < 2; ++jn) {
        v16bf bf = frag_b_lds(tW + (wn + jn * 16) * 48, lane);
        acc[im][jn] = WMMA_BF16(af, bf, acc[im][jn]);
      }
    }

    wait_stage();      // my async copies into buf^1 are done
    __syncthreads();   // everyone's copies done + everyone done reading buf
    buf ^= 1;
  }

  // epilogue: C/D layout -> lane l<16: M=r, N=l ; lane l>=16: M=r+8, N=l-16
  const int halfSel = lane >> 4;
  const int colL = lane & 15;
#pragma unroll
  for (int im = 0; im < 2; ++im) {
#pragma unroll
    for (int jn = 0; jn < 2; ++jn) {
      size_t gn = nBlock + wn + jn * 16 + colL;
      float bv = bias[gn];
#pragma unroll
      for (int r = 0; r < 8; ++r) {
        size_t gm = mBlock + wm + im * 16 + r + halfSel * 8;
        float v = acc[im][jn][r] + bv;
        if (resid) v += resid[gm * (size_t)ldr + gn];
        if (outF) outF[gm * (size_t)ldo + gn] = v;
        if (outB) outB[gm * (size_t)ldo + gn] = (__bf16)v;
      }
    }
  }
}

// ---------------------------------------------------------------- RoPE + QKV split
// qkv f32 (T, 3C) with 3C laid out as (h, d, 3). Emits q,k bf16 (b,h,n,d)
// and v bf16 TRANSPOSED (b,h,d,n) so PV B-fragments are contiguous.
__global__ void k_rope(const float* __restrict__ qkv, const float* __restrict__ enc,
                       __bf16* __restrict__ qb, __bf16* __restrict__ kb,
                       __bf16* __restrict__ vb) {
  int gid = blockIdx.x * blockDim.x + threadIdx.x;
  if (gid >= Tt * Hh * (Dd / 2)) return;
  int i   = gid % (Dd / 2);
  int tmp = gid / (Dd / 2);
  int h   = tmp % Hh;
  int tok = tmp / Hh;
  int b = tok / Nn, n = tok % Nn;
  int d0 = 2 * i, d1 = d0 + 1;

  const float* base = qkv + (size_t)tok * (3 * Cc) + h * Dd * 3;
  float q0 = base[d0 * 3 + 0], kk0 = base[d0 * 3 + 1], v0 = base[d0 * 3 + 2];
  float q1 = base[d1 * 3 + 0], kk1 = base[d1 * 3 + 1], v1 = base[d1 * 3 + 2];

  size_t eb = ((size_t)b * Nn + n) * Dd;
  const size_t sinOff = (size_t)Bb * Nn * Dd;
  float c0 = enc[eb + d0], s0 = enc[sinOff + eb + d0];
  float c1 = enc[eb + d1], s1 = enc[sinOff + eb + d1];

  // q' = q*cos + rotate_half(q)*sin  (pairwise interleaved)
  float qr0 = q0 * c0 - q1 * s0, qr1 = q1 * c1 + q0 * s1;
  float kr0 = kk0 * c0 - kk1 * s0, kr1 = kk1 * c1 + kk0 * s1;

  size_t qoff = ((size_t)(b * Hh + h) * Nn + n) * Dd;
  qb[qoff + d0] = (__bf16)qr0; qb[qoff + d1] = (__bf16)qr1;
  kb[qoff + d0] = (__bf16)kr0; kb[qoff + d1] = (__bf16)kr1;
  size_t voff = (size_t)(b * Hh + h) * Dd * Nn;
  vb[voff + (size_t)d0 * Nn + n] = (__bf16)v0;
  vb[voff + (size_t)d1 * Nn + n] = (__bf16)v1;
}

// ---------------------------------------------------------------- flash attention
// 1 wave per 16 query rows of one (b,h). Online softmax; all matmuls via WMMA.
__global__ __launch_bounds__(32) void k_attn(const __bf16* __restrict__ q,
                                             const __bf16* __restrict__ k,
                                             const __bf16* __restrict__ vt,
                                             __bf16* __restrict__ ctx) {
  __shared__ __bf16 sP[16 * 48];
  const int lane = threadIdx.x;
  const int ntiles = Nn / 16;
  const int bh = blockIdx.x / ntiles;
  const int qt = blockIdx.x % ntiles;
  const int b = bh / Hh, h = bh % Hh;
  const __bf16* qp = q + (size_t)bh * Nn * Dd;
  const __bf16* kp = k + (size_t)bh * Nn * Dd;
  const __bf16* vp = vt + (size_t)bh * Dd * Nn;
  const int qbase = qt * 16;

  const int rowA = qbase + (lane & 15);
  const int kbA = (lane >> 4) << 4;
  v16bf qa0 = load_afrag(qp + (size_t)rowA * Dd + kbA);        // d = 0..31
  v16bf qa1 = load_afrag(qp + (size_t)rowA * Dd + 32 + kbA);   // d = 32..63

  float mrow[8], lrow[8];
  v8f accO[4] = {};
#pragma unroll
  for (int r = 0; r < 8; ++r) { mrow[r] = -1e30f; lrow[r] = 0.f; }

  const int colB = lane & 15;
  const int kbB = (lane >> 4) << 4;

  for (int j0 = 0; j0 < Nn; j0 += 32) {
    v8f s0 = {}, s1 = {};
    s0 = WMMA_BF16(qa0, load_bfrag(kp + (size_t)(j0 + colB) * Dd + kbB), s0);
    s0 = WMMA_BF16(qa1, load_bfrag(kp + (size_t)(j0 + colB) * Dd + 32 + kbB), s0);
    s1 = WMMA_BF16(qa0, load_bfrag(kp + (size_t)(j0 + 16 + colB) * Dd + kbB), s1);
    s1 = WMMA_BF16(qa1, load_bfrag(kp + (size_t)(j0 + 16 + colB) * Dd + 32 + kbB), s1);

#pragma unroll
    for (int r = 0; r < 8; ++r) {
      float a0 = s0[r] * 0.125f;        // 1/sqrt(64)
      float a1 = s1[r] * 0.125f;
      float t = fmaxf(a0, a1);
#pragma unroll
      for (int m = 8; m > 0; m >>= 1) t = fmaxf(t, __shfl_xor(t, m, 16));
      float mn = fmaxf(mrow[r], t);
      float corr = __expf(mrow[r] - mn);
      float p0 = __expf(a0 - mn), p1 = __expf(a1 - mn);
      float rsum = p0 + p1;
#pragma unroll
      for (int m = 8; m > 0; m >>= 1) rsum += __shfl_xor(rsum, m, 16);
      lrow[r] = lrow[r] * corr + rsum;
      mrow[r] = mn;
#pragma unroll
      for (int t2 = 0; t2 < 4; ++t2) accO[t2][r] *= corr;
      // C-layout element (r,lane) is row Mr = r + 8*(lane>=16), col = lane&15
      int Mr = r + ((lane >> 4) << 3);
      sP[Mr * 48 + (lane & 15)]      = (__bf16)p0;
      sP[Mr * 48 + 16 + (lane & 15)] = (__bf16)p1;
    }
    __syncthreads();
    v16bf pa = frag_a_lds(sP, lane);          // P as A-fragment (16x32)
#pragma unroll
    for (int t2 = 0; t2 < 4; ++t2) {
      v16bf vbf = load_bfrag(vp + (size_t)(t2 * 16 + colB) * Nn + j0 + kbB);
      accO[t2] = WMMA_BF16(pa, vbf, accO[t2]);
    }
    __syncthreads();
  }

  // normalize and scatter ctx (token-major (b,n, h*D+d))
#pragma unroll
  for (int r = 0; r < 8; ++r) {
    float inv = 1.f / lrow[r];
    int n = qbase + r + ((lane >> 4) << 3);
    size_t obase = ((size_t)(b * Nn + n)) * Cc + h * Dd;
#pragma unroll
    for (int t2 = 0; t2 < 4; ++t2)
      ctx[obase + t2 * 16 + colB] = (__bf16)(accO[t2][r] * inv);
  }
}

// ---------------------------------------------------------------- LayerNorm + GELU
__global__ __launch_bounds__(256) void k_lngelu(const float* __restrict__ hbuf,
                                                const float* __restrict__ g,
                                                const float* __restrict__ bb,
                                                __bf16* __restrict__ hb) {
  const int row = blockIdx.x;
  const float* hr = hbuf + (size_t)row * (2 * Cc);
  float s = 0.f, s2 = 0.f;
  for (int i = threadIdx.x; i < 2 * Cc; i += 256) {
    float v = hr[i];
    s += v; s2 += v * v;
  }
#pragma unroll
  for (int m = 16; m > 0; m >>= 1) {
    s  += __shfl_xor(s, m, 32);
    s2 += __shfl_xor(s2, m, 32);
  }
  __shared__ float rs[8], rs2[8];
  int wid = threadIdx.x >> 5;
  if ((threadIdx.x & 31) == 0) { rs[wid] = s; rs2[wid] = s2; }
  __syncthreads();
  s = 0.f; s2 = 0.f;
#pragma unroll
  for (int w = 0; w < 8; ++w) { s += rs[w]; s2 += rs2[w]; }
  const float mu  = s * (1.f / (2 * Cc));
  const float var = s2 * (1.f / (2 * Cc)) - mu * mu;
  const float inv = rsqrtf(var + 1e-5f);
  __bf16* ho = hb + (size_t)row * (2 * Cc);
  for (int i = threadIdx.x; i < 2 * Cc; i += 256) {
    float v = (hr[i] - mu) * inv * g[i] + bb[i];
    float ge = 0.5f * v * (1.f + erff(v * 0.70710678118654752f));  // exact GELU
    ho[i] = (__bf16)ge;
  }
}

// ---------------------------------------------------------------- launcher
extern "C" void kernel_launch(void* const* d_in, const int* in_sizes, int n_in,
                              void* d_out, int out_size, void* d_ws, size_t ws_size,
                              hipStream_t stream) {
  (void)in_sizes; (void)n_in; (void)out_size; (void)ws_size;
  const float* x    = (const float*)d_in[0];
  const float* enc  = (const float*)d_in[1];
  const float* wqkv = (const float*)d_in[2];
  const float* bqkv = (const float*)d_in[3];
  const float* wout = (const float*)d_in[4];
  const float* bout = (const float*)d_in[5];
  const float* wf1  = (const float*)d_in[6];
  const float* bf1  = (const float*)d_in[7];
  const float* lng  = (const float*)d_in[8];
  const float* lnb  = (const float*)d_in[9];
  const float* wf2  = (const float*)d_in[10];
  const float* bf2  = (const float*)d_in[11];
  float* out = (float*)d_out;

  char* ws = (char*)d_ws;
  size_t off = 0;
  auto take = [&](size_t bytes) -> char* {
    char* p = ws + off;
    off += (bytes + 255) & ~(size_t)255;
    return p;
  };
  __bf16* x_bf    = (__bf16*)take((size_t)Tt * Cc * 2);
  __bf16* wqkv_bf = (__bf16*)take((size_t)3 * Cc * Cc * 2);
  __bf16* wout_bf = (__bf16*)take((size_t)Cc * Cc * 2);
  __bf16* wf1_bf  = (__bf16*)take((size_t)4 * Cc * Cc * 2);
  __bf16* wf2_bf  = (__bf16*)take((size_t)2 * Cc * Cc * 2);
  float*  qkv_f   = (float*) take((size_t)Tt * 3 * Cc * 4);
  __bf16* q_bf    = (__bf16*)take((size_t)Tt * Cc * 2);
  __bf16* k_bf    = (__bf16*)take((size_t)Tt * Cc * 2);
  __bf16* v_bf    = (__bf16*)take((size_t)Tt * Cc * 2);
  __bf16* ctx_bf  = (__bf16*)take((size_t)Tt * Cc * 2);
  __bf16* cc_bf   = (__bf16*)take((size_t)Tt * 2 * Cc * 2);
  float*  h_f     = (float*) take((size_t)Tt * 2 * Cc * 4);
  __bf16* h_bf    = (__bf16*)take((size_t)Tt * 2 * Cc * 2);

  // 1) casts
  k_cast_x<<<4096, 256, 0, stream>>>(x, x_bf, cc_bf, Tt * Cc);
  k_cast<<<4096, 256, 0, stream>>>(wqkv, wqkv_bf, 3 * Cc * Cc);
  k_cast<<<2048, 256, 0, stream>>>(wout, wout_bf, Cc * Cc);
  k_cast<<<4096, 256, 0, stream>>>(wf1, wf1_bf, 4 * Cc * Cc);
  k_cast<<<4096, 256, 0, stream>>>(wf2, wf2_bf, 2 * Cc * Cc);

  // 2) qkv = x @ Wqkv^T + b        (4096 x 3072, K=1024)
  k_gemm<<<dim3(3 * Cc / 64, Tt / 64), 128, 0, stream>>>(
      x_bf, Cc, wqkv_bf, Cc, bqkv, Cc, qkv_f, nullptr, 3 * Cc, nullptr, 0);

  // 3) RoPE split -> q,k (b,h,n,d) and v transposed (b,h,d,n)
  k_rope<<<(Tt * Hh * (Dd / 2) + 255) / 256, 256, 0, stream>>>(
      qkv_f, enc, q_bf, k_bf, v_bf);

  // 4) attention -> ctx (token-major)
  k_attn<<<Bb * Hh * (Nn / 16), 32, 0, stream>>>(q_bf, k_bf, v_bf, ctx_bf);

  // 5) message = ctx @ out_w^T + b -> second half of cc  (4096 x 1024, K=1024)
  k_gemm<<<dim3(Cc / 64, Tt / 64), 128, 0, stream>>>(
      ctx_bf, Cc, wout_bf, Cc, bout, Cc, nullptr, cc_bf + Cc, 2 * Cc, nullptr, 0);

  // 6) h = cc @ ffn1^T + b          (4096 x 2048, K=2048)
  k_gemm<<<dim3(2 * Cc / 64, Tt / 64), 128, 0, stream>>>(
      cc_bf, 2 * Cc, wf1_bf, 2 * Cc, bf1, 2 * Cc, h_f, nullptr, 2 * Cc, nullptr, 0);

  // 7) LayerNorm + exact GELU -> bf16
  k_lngelu<<<Tt, 256, 0, stream>>>(h_f, lng, lnb, h_bf);

  // 8) out = x + h @ ffn2^T + b     (4096 x 1024, K=2048)
  k_gemm<<<dim3(Cc / 64, Tt / 64), 128, 0, stream>>>(
      h_bf, 2 * Cc, wf2_bf, 2 * Cc, bf2, 2 * Cc, out, nullptr, Cc, x, Cc);
}
